// Attention_20229295964624
// MI455X (gfx1250) — matrix operand order repaired and checked by reading, not compiled, not run on hardware
//
#include <hip/hip_runtime.h>
#include <hip/hip_bf16.h>

// ---------------------------------------------------------------------------
// Types for CDNA5 WMMA (wave32): v_wmma_f32_16x16x32_bf16
// ---------------------------------------------------------------------------
typedef __attribute__((ext_vector_type(16))) __bf16 v16bf;
typedef __attribute__((ext_vector_type(8)))  float  v8f;

union Frag {
    v16bf v;
    uint4 q[2];
};

__device__ __forceinline__ unsigned short f2bf(float f) {
    unsigned int x = __float_as_uint(f);
    unsigned int r = (x + 0x7FFFu + ((x >> 16) & 1u)) >> 16; // RNE
    return (unsigned short)r;
}
__device__ __forceinline__ float bf2f(unsigned short u) {
    return __uint_as_float(((unsigned int)u) << 16);
}

#define WMMA_BF16(A, B, C) \
    __builtin_amdgcn_wmma_f32_16x16x32_bf16(false, (A), false, (B), (short)0, (C), false, false)

// Problem constants
static const int Bk = 4, HWk = 2048, Ck = 768, Hk = 8, HDk = 96;

// ---------------------------------------------------------------------------
// f32 -> bf16 convert
// ---------------------------------------------------------------------------
__global__ void f32_to_bf16_kernel(const float* __restrict__ in,
                                   unsigned short* __restrict__ out, long n) {
    long i = (long)blockIdx.x * blockDim.x + threadIdx.x;
    if (i < n) out[i] = f2bf(in[i]);
}

// ---------------------------------------------------------------------------
// Generic bf16 GEMM:  C[M,N] = A[M,K] * W[N,K]^T
// MODE 0: write bf16 row-major (Cb).  MODE 1: write f32 + bias (Cf).
// Block: 256 threads = 8 waves (4x2), 128x64 tile, K-step 32,
// double-buffered LDS (one barrier per K-step) + global_prefetch 2 steps ahead.
// ---------------------------------------------------------------------------
#define LDP 40   // LDS row pitch in bf16 (80B, 16B-aligned)

template <int MODE>
__global__ __launch_bounds__(256) void gemm_bf16_kernel(
    const unsigned short* __restrict__ A, const unsigned short* __restrict__ W,
    float* __restrict__ Cf, unsigned short* __restrict__ Cb,
    const float* __restrict__ bias, int M, int N, int K) {
    __shared__ unsigned short As[2][128 * LDP];
    __shared__ unsigned short Bs[2][64 * LDP];

    const int tid  = threadIdx.x;
    const int lane = tid & 31;
    const int wv   = tid >> 5;          // 8 waves
    const int wm   = wv >> 1, wn = wv & 1;   // 4 x 2
    const int half = lane >> 4, l16 = lane & 15;
    const long mBase = (long)blockIdx.y * 128;
    const long nBase = (long)blockIdx.x * 64;

    auto loadTiles = [&](int k0, int buf) {
#pragma unroll
        for (int i = 0; i < 2; ++i) {       // A: 128 rows x 32 cols
            int lin = tid + i * 256;
            int r = lin >> 2, c = (lin & 3) << 3;
            *(uint4*)&As[buf][r * LDP + c] = *(const uint4*)&A[(mBase + r) * K + k0 + c];
        }
        {                                   // B: 64 rows x 32 cols
            int r = tid >> 2, c = (tid & 3) << 3;
            *(uint4*)&Bs[buf][r * LDP + c] = *(const uint4*)&W[(nBase + r) * K + k0 + c];
        }
    };

    v8f acc[2][2];
#pragma unroll
    for (int mi = 0; mi < 2; ++mi)
#pragma unroll
        for (int ni = 0; ni < 2; ++ni) acc[mi][ni] = (v8f){};

    const int nk = K / 32;
    loadTiles(0, 0);

    for (int t = 0; t < nk; ++t) {
        __syncthreads();
        if (t + 1 < nk) loadTiles((t + 1) * 32, (t + 1) & 1);
        if (t + 2 < nk) {                   // speculative prefetch -> global_prefetch_b8
            int k2 = (t + 2) * 32;
#pragma unroll
            for (int i = 0; i < 2; ++i) {
                int lin = tid + i * 256;
                int r = lin >> 2, c = (lin & 3) << 3;
                __builtin_prefetch(&A[(mBase + r) * K + k2 + c], 0, 0);
            }
            __builtin_prefetch(&W[(nBase + (tid >> 2)) * K + k2 + ((tid & 3) << 3)], 0, 0);
        }
        const int cur = t & 1;

        Frag a[2], b[2];
#pragma unroll
        for (int mi = 0; mi < 2; ++mi) {
            const unsigned short* p = &As[cur][(wm * 32 + mi * 16 + l16) * LDP + 8 * half];
            a[mi].q[0] = *(const uint4*)p;
            a[mi].q[1] = *(const uint4*)(p + 16);
        }
#pragma unroll
        for (int ni = 0; ni < 2; ++ni) {
            const unsigned short* p = &Bs[cur][(wn * 32 + ni * 16 + l16) * LDP + 8 * half];
            b[ni].q[0] = *(const uint4*)p;
            b[ni].q[1] = *(const uint4*)(p + 16);
        }
#pragma unroll
        for (int mi = 0; mi < 2; ++mi)
#pragma unroll
            for (int ni = 0; ni < 2; ++ni)
                acc[mi][ni] = WMMA_BF16(a[mi].v, b[ni].v, acc[mi][ni]);
    }

#pragma unroll
    for (int mi = 0; mi < 2; ++mi)
#pragma unroll
        for (int ni = 0; ni < 2; ++ni)
#pragma unroll
            for (int j = 0; j < 8; ++j) {
                long row = mBase + wm * 32 + mi * 16 + half * 8 + j;
                long col = nBase + wn * 32 + ni * 16 + l16;
                float v = acc[mi][ni][j];
                if (MODE == 0) Cb[row * N + col] = f2bf(v);
                else           Cf[row * N + col] = v + bias[col];
            }
}

// ---------------------------------------------------------------------------
// RoPE (interleaved rotate-half) + head split.
// qkv bf16 [B*HW, 2304] -> Q,K [B,H,HW,HD] bf16 ; V transposed [B,H,HD,HW].
// One thread per (b,h,pos,pair).
// ---------------------------------------------------------------------------
__global__ void rope_split_kernel(const unsigned short* __restrict__ qkv,
                                  const float* __restrict__ cosb,
                                  const float* __restrict__ sinb,
                                  unsigned short* __restrict__ Qo,
                                  unsigned short* __restrict__ Ko,
                                  unsigned short* __restrict__ Vt) {
    long t = (long)blockIdx.x * blockDim.x + threadIdx.x;
    const long total = (long)Bk * Hk * HWk * (HDk / 2);
    if (t >= total) return;
    int i = (int)(t % 48); long r = t / 48;
    int pos = (int)(r % HWk); r /= HWk;
    int h = (int)(r % Hk); int b = (int)(r / Hk);
    int d0 = 2 * i;

    long base = ((long)(b * HWk + pos)) * (3 * Ck) + h * HDk + d0;
    float q0 = bf2f(qkv[base]),        q1 = bf2f(qkv[base + 1]);
    float k0 = bf2f(qkv[base + Ck]),   k1 = bf2f(qkv[base + Ck + 1]);
    float v0 = bf2f(qkv[base + 2*Ck]), v1 = bf2f(qkv[base + 2*Ck + 1]);

    float c0 = cosb[pos * HDk + d0], c1 = cosb[pos * HDk + d0 + 1];
    float s0 = sinb[pos * HDk + d0], s1 = sinb[pos * HDk + d0 + 1];
    // out = t*cos + rot(t)*sin ; rot[2i] = -t[2i+1], rot[2i+1] = t[2i]
    float qo0 = q0 * c0 - q1 * s0, qo1 = q1 * c1 + q0 * s1;
    float ko0 = k0 * c0 - k1 * s0, ko1 = k1 * c1 + k0 * s1;

    long qoff = ((long)((b * Hk + h) * HWk + pos)) * HDk + d0;
    Qo[qoff] = f2bf(qo0); Qo[qoff + 1] = f2bf(qo1);
    Ko[qoff] = f2bf(ko0); Ko[qoff + 1] = f2bf(ko1);
    long voff = ((long)((b * Hk + h) * HDk + d0)) * HWk + pos;
    Vt[voff] = f2bf(v0); Vt[voff + HWk] = f2bf(v1);
}

// ---------------------------------------------------------------------------
// Flash attention: block = 128 threads (4 waves), one (b,h,64-row q-tile).
// Wave w owns rows [16w,16w+16). Online softmax over 64-wide KV chunks.
// S f32 tile in LDS is overwritten in place by P (bf16) during softmax.
// ---------------------------------------------------------------------------
#define QP 104   // Q/K LDS pitch (bf16), 208B
#define VP 72    // Vt LDS pitch (bf16), 144B
#define SP 68    // S LDS pitch (f32), 272B ; P bf16 pitch = 136 shorts

__global__ __launch_bounds__(128) void attn_kernel(
    const unsigned short* __restrict__ Qg, const unsigned short* __restrict__ Kg,
    const unsigned short* __restrict__ Vtg, unsigned short* __restrict__ Og) {
    __shared__ unsigned short Qs[64 * QP];
    __shared__ unsigned short Ks[64 * QP];
    __shared__ unsigned short Vs[96 * VP];
    __shared__ float Ss[64 * SP];
    __shared__ float mrow[64], lrow[64], arow[64];

    const int tid  = threadIdx.x;
    const int lane = tid & 31, wv = tid >> 5;
    const int half = lane >> 4, l16 = lane & 15;
    const int qt = blockIdx.x, h = blockIdx.y, b = blockIdx.z;
    const long bh = (long)b * Hk + h;

    const unsigned short* Qp = Qg + (bh * HWk + (long)qt * 64) * HDk;
    const unsigned short* Kp = Kg + bh * HWk * HDk;
    const unsigned short* Vp = Vtg + bh * HDk * HWk;

#pragma unroll
    for (int i = 0; i < 6; ++i) {                 // Q tile: 64 x 96 bf16
        int lin = tid + i * 128;
        int r = lin / 12, c = (lin % 12) * 8;
        *(uint4*)&Qs[r * QP + c] = *(const uint4*)&Qp[r * HDk + c];
    }
    if (tid < 64) { mrow[tid] = -3.0e38f; lrow[tid] = 0.0f; }

    v8f o[6];
#pragma unroll
    for (int ni = 0; ni < 6; ++ni) o[ni] = (v8f){};

    for (int kc = 0; kc < HWk / 64; ++kc) {
        __syncthreads();
        const unsigned short* Kc = Kp + (long)kc * 64 * HDk;
#pragma unroll
        for (int i = 0; i < 6; ++i) {             // K chunk: 64 x 96
            int lin = tid + i * 128;
            int r = lin / 12, c = (lin % 12) * 8;
            *(uint4*)&Ks[r * QP + c] = *(const uint4*)&Kc[r * HDk + c];
        }
#pragma unroll
        for (int i = 0; i < 6; ++i) {             // Vt chunk: 96 x 64
            int lin = tid + i * 128;
            int r = lin / 8, c = (lin % 8) * 8;
            *(uint4*)&Vs[r * VP + c] = *(const uint4*)&Vp[(long)r * HWk + kc * 64 + c];
        }
        if (kc + 1 < HWk / 64) {                  // prefetch next chunk into GL2/WGP$
            const unsigned short* Kn = Kc + 64 * HDk;
#pragma unroll
            for (int i = 0; i < 6; ++i) {
                int lin = tid + i * 128;
                int r = lin / 12, c = (lin % 12) * 8;
                __builtin_prefetch(&Kn[r * HDk + c], 0, 0);
            }
#pragma unroll
            for (int i = 0; i < 6; ++i) {
                int lin = tid + i * 128;
                int r = lin / 8, c = (lin % 8) * 8;
                __builtin_prefetch(&Vp[(long)r * HWk + (kc + 1) * 64 + c], 0, 0);
            }
        }
        __syncthreads();

        // ---- S = Q K^T (HD=96 -> 3 k-steps) ----
        v8f s[4];
#pragma unroll
        for (int ni = 0; ni < 4; ++ni) s[ni] = (v8f){};
#pragma unroll
        for (int ks = 0; ks < 3; ++ks) {
            Frag a;
            const unsigned short* pa = &Qs[(wv * 16 + l16) * QP + ks * 32 + 8 * half];
            a.q[0] = *(const uint4*)pa; a.q[1] = *(const uint4*)(pa + 16);
#pragma unroll
            for (int ni = 0; ni < 4; ++ni) {
                Frag bf;
                const unsigned short* pb = &Ks[(ni * 16 + l16) * QP + ks * 32 + 8 * half];
                bf.q[0] = *(const uint4*)pb; bf.q[1] = *(const uint4*)(pb + 16);
                s[ni] = WMMA_BF16(a.v, bf.v, s[ni]);
            }
        }
        const float scale = 0.1020620726159658f;  // 96^-0.5
#pragma unroll
        for (int ni = 0; ni < 4; ++ni)
#pragma unroll
            for (int j = 0; j < 8; ++j)
                Ss[(wv * 16 + half * 8 + j) * SP + ni * 16 + l16] = s[ni][j] * scale;
        __syncthreads();

        // ---- online softmax, one thread per row ----
        if (tid < 64) {
            int r = tid;
            float mold = mrow[r];
            float mx = mold;
            for (int c = 0; c < 64; ++c) mx = fmaxf(mx, Ss[r * SP + c]);
            float alpha = __expf(mold - mx);
            float lsum = 0.0f;
            unsigned short* Pp = (unsigned short*)Ss;
            for (int c = 0; c < 64; ++c) {
                float p = __expf(Ss[r * SP + c] - mx);
                lsum += p;
                Pp[r * (2 * SP) + c] = f2bf(p);   // overwrite S row in place
            }
            mrow[r] = mx;
            lrow[r] = lrow[r] * alpha + lsum;
            arow[r] = alpha;
        }
        __syncthreads();

        // ---- O = O*alpha + P * V ----
        float al[8];
#pragma unroll
        for (int j = 0; j < 8; ++j) al[j] = arow[wv * 16 + half * 8 + j];
#pragma unroll
        for (int ni = 0; ni < 6; ++ni)
#pragma unroll
            for (int j = 0; j < 8; ++j) o[ni][j] *= al[j];

        const unsigned short* Pp = (const unsigned short*)Ss;
#pragma unroll
        for (int ks = 0; ks < 2; ++ks) {
            Frag a;
            const unsigned short* pa = &Pp[(wv * 16 + l16) * (2 * SP) + ks * 32 + 8 * half];
            a.q[0] = *(const uint4*)pa; a.q[1] = *(const uint4*)(pa + 16);
#pragma unroll
            for (int ni = 0; ni < 6; ++ni) {
                Frag bf;
                const unsigned short* pb = &Vs[(ni * 16 + l16) * VP + ks * 32 + 8 * half];
                bf.q[0] = *(const uint4*)pb; bf.q[1] = *(const uint4*)(pb + 16);
                o[ni] = WMMA_BF16(a.v, bf.v, o[ni]);
            }
        }
    }

    // ---- normalize and store: attnOut [B, HW, H*HD] bf16 ----
    float inv[8];
#pragma unroll
    for (int j = 0; j < 8; ++j) inv[j] = 1.0f / lrow[wv * 16 + half * 8 + j];
#pragma unroll
    for (int ni = 0; ni < 6; ++ni)
#pragma unroll
        for (int j = 0; j < 8; ++j) {
            int m = wv * 16 + half * 8 + j;
            int d = ni * 16 + l16;
            long off = ((long)(b * HWk + qt * 64 + m)) * Ck + h * HDk + d;
            Og[off] = f2bf(o[ni][j] * inv[j]);
        }
}

// ---------------------------------------------------------------------------
// Host-side launch
// ---------------------------------------------------------------------------
extern "C" void kernel_launch(void* const* d_in, const int* in_sizes, int n_in,
                              void* d_out, int out_size, void* d_ws, size_t ws_size,
                              hipStream_t stream) {
    const float* x     = (const float*)d_in[0];
    const float* cosb  = (const float*)d_in[1];
    const float* sinb  = (const float*)d_in[2];
    const float* Wqkv  = (const float*)d_in[3];
    const float* Wproj = (const float*)d_in[4];
    const float* bproj = (const float*)d_in[5];
    float* out = (float*)d_out;

    const long M = (long)Bk * HWk;      // 8192
    const long N1 = 3 * Ck;             // 2304
    const long Kdim = Ck;               // 768
    auto align256 = [](size_t v) { return (v + 255) & ~(size_t)255; };

    char* w = (char*)d_ws;
    size_t off = 0;
    unsigned short* xb = (unsigned short*)(w + off); off = align256(off + (size_t)M * Ck * 2);
    unsigned short* wqkvb = (unsigned short*)(w + off); off = align256(off + (size_t)N1 * Ck * 2);
    unsigned short* wprojb = (unsigned short*)(w + off); off = align256(off + (size_t)Ck * Ck * 2);
    unsigned short* qkvb = (unsigned short*)(w + off); off = align256(off + (size_t)M * N1 * 2);
    unsigned short* Qw = (unsigned short*)(w + off); off = align256(off + (size_t)M * Ck * 2);
    unsigned short* Kw = (unsigned short*)(w + off); off = align256(off + (size_t)M * Ck * 2);
    unsigned short* Vtw = (unsigned short*)(w + off); off = align256(off + (size_t)M * Ck * 2);
    unsigned short* aow = (unsigned short*)(w + off); off = align256(off + (size_t)M * Ck * 2);
    if (off > ws_size) return;

    // 1) converts
    {
        long n = M * Ck;
        f32_to_bf16_kernel<<<(n + 255) / 256, 256, 0, stream>>>(x, xb, n);
        n = N1 * Ck;
        f32_to_bf16_kernel<<<(n + 255) / 256, 256, 0, stream>>>(Wqkv, wqkvb, n);
        n = (long)Ck * Ck;
        f32_to_bf16_kernel<<<(n + 255) / 256, 256, 0, stream>>>(Wproj, wprojb, n);
    }
    // 2) QKV GEMM (bf16 out)
    gemm_bf16_kernel<0><<<dim3(N1 / 64, M / 128), 256, 0, stream>>>(
        xb, wqkvb, nullptr, qkvb, nullptr, (int)M, (int)N1, (int)Kdim);
    // 3) RoPE + head split (+ V transpose)
    {
        long n = (long)Bk * Hk * HWk * (HDk / 2);
        rope_split_kernel<<<(n + 255) / 256, 256, 0, stream>>>(qkvb, cosb, sinb, Qw, Kw, Vtw);
    }
    // 4) flash attention
    attn_kernel<<<dim3(HWk / 64, Hk, Bk), 128, 0, stream>>>(Qw, Kw, Vtw, aow);
    // 5) projection GEMM (f32 out + bias)
    gemm_bf16_kernel<1><<<dim3(Ck / 64, M / 128), 256, 0, stream>>>(
        aow, wprojb, out, nullptr, bproj, (int)M, (int)Ck, (int)Kdim);
}
